// RNNLayer_317827580243
// MI455X (gfx1250) — matrix operand order repaired and checked by reading, not compiled, not run on hardware
//
#include <hip/hip_runtime.h>
#include <hip/hip_bf16.h>
#include <math.h>

// ---------------- problem constants ----------------
#define B_   16
#define T_   1000
#define F_   512
#define H_   512
#define TT_  2000            // 2*T
#define M1_  (TT_*B_)        // 32000 rows of pre-activation GEMM (m = t*16 + b)
#define M2_  (B_*T_)         // 16000 rows of projection GEMM (m = b*T + t)

typedef __attribute__((ext_vector_type(16))) _Float16 v16h;
typedef __attribute__((ext_vector_type(8)))  float    v8f;
typedef __attribute__((ext_vector_type(4)))  int      v4i;

union HFrag { v16h h; uint4 u[2]; };

// ---------------- gfx1250 async global->LDS path (guarded; falls back to ld/st) ----------------
#if __has_builtin(__builtin_amdgcn_global_load_async_to_lds_b128) && \
    __has_builtin(__builtin_amdgcn_s_wait_asynccnt)
#define HAS_ASYNC_LDS 1
__device__ __forceinline__ void async_g2l_b128(const void* g, void* l) {
    __builtin_amdgcn_global_load_async_to_lds_b128(
        (__attribute__((address_space(1))) v4i*)g,
        (__attribute__((address_space(3))) v4i*)l, 0, 0);
}
#define ASYNC_DRAIN() __builtin_amdgcn_s_wait_asynccnt(0)
#else
#define HAS_ASYNC_LDS 0
#define ASYNC_DRAIN()
#endif

// ---------------- workspace layout ----------------
static constexpr size_t ALGN(size_t x) { return (x + 255) & ~size_t(255); }
static constexpr size_t OFF_XC    = 0;                                   // xc_f16 [M1][512]
static constexpr size_t SZ_XC     = (size_t)M1_ * 512 * 2;
static constexpr size_t OFF_WC    = ALGN(OFF_XC + SZ_XC);                // Wc_f16 [1024][512]
static constexpr size_t SZ_WC     = (size_t)1024 * 512 * 2;
static constexpr size_t OFF_SCALE = ALGN(OFF_WC + SZ_WC);                // f32 [1024]
static constexpr size_t OFF_SHIFT = OFF_SCALE + 4096;                    // f32 [1024]
static constexpr size_t OFF_U16   = ALGN(OFF_SHIFT + 4096);              // U_f16 [1024][512]
static constexpr size_t SZ_U16    = (size_t)1024 * 512 * 2;
static constexpr size_t OFF_PJW   = ALGN(OFF_U16 + SZ_U16);              // pjW_f16 [1024][1024]
static constexpr size_t SZ_PJW    = (size_t)1024 * 1024 * 2;
static constexpr size_t OFF_WZ    = ALGN(OFF_PJW + SZ_PJW);              // WZ_f16 [M1][512]
static constexpr size_t SZ_WZH    = (size_t)M1_ * 512 * 2;
static constexpr size_t OFF_WH    = ALGN(OFF_WZ + SZ_WZH);               // WH_f16 [M1][512]
static constexpr size_t OFF_HS    = ALGN(OFF_WH + SZ_WZH);               // HS f32 [M1][512]
static constexpr size_t SZ_HS     = (size_t)M1_ * 512 * 4;
static constexpr size_t OFF_LN    = ALGN(OFF_HS + SZ_HS);                // ln_f16 [M2][1024]
static constexpr size_t SZ_LN     = (size_t)M2_ * 1024 * 2;
static constexpr size_t OFF_HBF   = ALGN(OFF_LN + SZ_LN);                // f16 [2][16][512]
static constexpr size_t OFF_CNT   = ALGN(OFF_HBF + (size_t)2*16*512*2);  // sync counter

// ---------------- prep kernels ----------------
__global__ void k_prep_wc(const float* __restrict__ Wz, const float* __restrict__ Wh,
                          _Float16* __restrict__ Wc) {
    int i = blockIdx.x * 256 + threadIdx.x;
    if (i >= 1024 * 512) return;
    int j = i >> 9, f = i & 511;
    float v = (j < 512) ? Wz[j * 512 + f] : Wh[(j - 512) * 512 + f];
    Wc[i] = (_Float16)v;
}

__global__ void k_prep_scale(const float* bz, const float* bh,
                             const float* zg, const float* zb, const float* zm, const float* zv,
                             const float* hg, const float* hb, const float* hm, const float* hv,
                             float* scale, float* shift) {
    int j = blockIdx.x * 256 + threadIdx.x;
    if (j >= 1024) return;
    if (j < 512) {
        float s = zg[j] * rsqrtf(zv[j] + 1e-5f);
        scale[j] = s; shift[j] = (bz[j] - zm[j]) * s + zb[j];
    } else {
        int k = j - 512;
        float s = hg[k] * rsqrtf(hv[k] + 1e-5f);
        scale[j] = s; shift[j] = (bh[k] - hm[k]) * s + hb[k];
    }
}

__global__ void k_f32_to_f16(const float* __restrict__ src, _Float16* __restrict__ dst, int n) {
    int i = blockIdx.x * 256 + threadIdx.x;
    if (i < n) dst[i] = (_Float16)src[i];
}

// x_cat row m = t*16 + b ; t<T: x[b][t] ; t>=T: x[15-b][t-T]   (x[::-1] flips batch)
__global__ void k_prep_xc(const float* __restrict__ x, _Float16* __restrict__ xc) {
    long i = (long)blockIdx.x * 256 + threadIdx.x;
    if (i >= (long)M1_ * 512) return;
    int f = (int)(i & 511);
    long m = i >> 9;
    int b = (int)(m & 15), t = (int)(m >> 4);
    float v = (t < T_) ? x[((size_t)b * T_ + t) * 512 + f]
                       : x[((size_t)(15 - b) * T_ + (t - T_)) * 512 + f];
    xc[i] = (_Float16)v;
}

__global__ void k_zero(unsigned* p) { *p = 0u; }

// ---------------- GEMM 1: pre-activations (M1 x 1024, K=512), BN folded ----------------
__global__ __launch_bounds__(256, 1)
void k_gemm1(const _Float16* __restrict__ Ag, const _Float16* __restrict__ Wg,
             const float* __restrict__ scale, const float* __restrict__ shift,
             _Float16* __restrict__ WZ, _Float16* __restrict__ WH) {
    __shared__ _Float16 As[128][40];
    __shared__ _Float16 Bs[128][40];
    const int bm = blockIdx.x, bn = blockIdx.y;
    const int tid = threadIdx.x, lane = tid & 31, wave = tid >> 5;
    const int mw = wave >> 1, nw = wave & 1;     // 4 x 2 wave grid
    const int kh = lane >> 4, ln15 = lane & 15;

    v8f zero = {0.f,0.f,0.f,0.f,0.f,0.f,0.f,0.f};
    v8f acc[2][4];
#pragma unroll
    for (int i = 0; i < 2; ++i)
#pragma unroll
        for (int j = 0; j < 4; ++j) acc[i][j] = zero;

    const int lr = tid >> 1, lp = (tid & 1) * 16;      // staging: 2 threads per row
    const _Float16* agp = Ag + (size_t)(bm * 128 + lr) * 512 + lp;
    const _Float16* bgp = Wg + (size_t)(bn * 128 + lr) * 512 + lp;

    for (int k0 = 0; k0 < 512; k0 += 32) {
        const uint4* asrc = (const uint4*)(agp + k0);
        const uint4* bsrc = (const uint4*)(bgp + k0);
#if HAS_ASYNC_LDS
        async_g2l_b128(asrc + 0, &As[lr][lp]);
        async_g2l_b128(asrc + 1, &As[lr][lp + 8]);
        async_g2l_b128(bsrc + 0, &Bs[lr][lp]);
        async_g2l_b128(bsrc + 1, &Bs[lr][lp + 8]);
#else
        *(uint4*)&As[lr][lp]     = asrc[0];
        *(uint4*)&As[lr][lp + 8] = asrc[1];
        *(uint4*)&Bs[lr][lp]     = bsrc[0];
        *(uint4*)&Bs[lr][lp + 8] = bsrc[1];
#endif
        if (k0 + 32 < 512) {                            // gfx1250 global_prefetch
            __builtin_prefetch(agp + k0 + 32, 0, 1);
            __builtin_prefetch(bgp + k0 + 32, 0, 1);
        }
        ASYNC_DRAIN();
        __syncthreads();

        HFrag a[2], b[4];
#pragma unroll
        for (int i = 0; i < 2; ++i) {
            const char* p = (const char*)As + (size_t)(mw * 32 + i * 16 + ln15) * 80;
            a[i].u[0] = *(const uint4*)(p + kh * 16);
            a[i].u[1] = *(const uint4*)(p + 32 + kh * 16);
        }
#pragma unroll
        for (int j = 0; j < 4; ++j) {
            const char* p = (const char*)Bs + (size_t)(nw * 64 + j * 16 + ln15) * 80;
            b[j].u[0] = *(const uint4*)(p + kh * 16);
            b[j].u[1] = *(const uint4*)(p + 32 + kh * 16);
        }
#pragma unroll
        for (int i = 0; i < 2; ++i)
#pragma unroll
            for (int j = 0; j < 4; ++j)
                acc[i][j] = __builtin_amdgcn_wmma_f32_16x16x32_f16(
                    false, a[i].h, false, b[j].h, (short)0, acc[i][j], false, false);
        __syncthreads();
    }
    // epilogue: BN affine, split columns into WZ / WH
#pragma unroll
    for (int i = 0; i < 2; ++i)
#pragma unroll
        for (int j = 0; j < 4; ++j) {
            int n = bn * 128 + nw * 64 + j * 16 + ln15;
            float sc = scale[n], sh = shift[n];
#pragma unroll
            for (int r = 0; r < 8; ++r) {
                int m = bm * 128 + mw * 32 + i * 16 + kh * 8 + r;
                float v = acc[i][j][r] * sc + sh;
                if (n < 512) WZ[(size_t)m * 512 + n] = (_Float16)v;
                else         WH[(size_t)m * 512 + (n - 512)] = (_Float16)v;
            }
        }
}

// ---------------- recurrence: 8 blocks, U resident in VGPRs ----------------
__global__ __launch_bounds__(256, 1)
void k_rnn(const _Float16* __restrict__ Ug, const _Float16* __restrict__ WZ,
           const _Float16* __restrict__ WH, float* __restrict__ HS,
           _Float16* __restrict__ Hbf, unsigned* __restrict__ cnt) {
#if __has_builtin(__builtin_amdgcn_s_wait_tensorcnt)
    __builtin_amdgcn_s_wait_tensorcnt(0);
#endif
    const int g = blockIdx.x;                 // 0..7 -> h columns [g*64, g*64+64)
    const int tid = threadIdx.x, lane = tid & 31, wave = tid >> 5;
    const int col0 = g * 64;
    const int ct = wave >> 1;                 // col-tile 0..3 within slice
    const int kind = wave & 1;                // 0 = uz rows of U, 1 = uh rows
    const int kh = lane >> 4, ln15 = lane & 15;

    __shared__ _Float16 hstage[16][520];      // full h (f16), padded rows
    __shared__ float    utile[2][4][16][16];  // [kind][ct][m][n]
    __shared__ float    hloc[16][64];         // this block's f32 h slice

    // Preload U slice as WMMA B-fragments: 16 frags x 8 VGPRs = 128 VGPRs, stays resident.
    HFrag uf[16];
    {
        int j = (kind ? 512 : 0) + col0 + ct * 16 + ln15;
        const char* base = (const char*)(Ug + (size_t)j * 512);
#pragma unroll
        for (int kt = 0; kt < 16; ++kt) {
            uf[kt].u[0] = *(const uint4*)(base + kt * 64 + kh * 16);
            uf[kt].u[1] = *(const uint4*)(base + kt * 64 + 32 + kh * 16);
        }
    }
    for (int i = tid; i < 16 * 64;  i += 256) hloc[i >> 6][i & 63] = 0.f;
    for (int i = tid; i < 16 * 520; i += 256) (&hstage[0][0])[i] = (_Float16)0.f;
    __syncthreads();

    v8f zero = {0.f,0.f,0.f,0.f,0.f,0.f,0.f,0.f};

    for (int t = 0; t < TT_; ++t) {
        // u-slice = h @ U^T for this wave's 16 columns (uz or uh)
        v8f acc = zero;
        {
            const char* ab = (const char*)&hstage[ln15][0];
#pragma unroll
            for (int kt = 0; kt < 16; ++kt) {
                HFrag a;
                a.u[0] = *(const uint4*)(ab + kt * 64 + kh * 16);
                a.u[1] = *(const uint4*)(ab + kt * 64 + 32 + kh * 16);
                acc = __builtin_amdgcn_wmma_f32_16x16x32_f16(
                    false, a.h, false, uf[kt].h, (short)0, acc, false, false);
            }
        }
#pragma unroll
        for (int r = 0; r < 8; ++r) utile[kind][ct][kh * 8 + r][ln15] = acc[r];
        __syncthreads();

        // elementwise LiGRU update for 4 columns per thread
        {
            int b = tid >> 4, c4 = (tid & 15) * 4;
            size_t rowoff = ((size_t)t * 16 + b) * 512 + col0 + c4;
#pragma unroll
            for (int i = 0; i < 4; ++i) {
                int cc = c4 + i;
                float uz = utile[0][cc >> 4][b][cc & 15];
                float uh = utile[1][cc >> 4][b][cc & 15];
                float wz = (float)WZ[rowoff + i];
                float wh = (float)WH[rowoff + i];
                float z  = 1.f / (1.f + __expf(-(wz + uz)));
                float hc = wh + uh; hc = hc > 0.f ? hc : 0.f;
                float ho = hloc[b][cc];
                float hn = z * ho + (1.f - z) * hc;
                hloc[b][cc] = hn;
                HS[rowoff + i] = hn;
                Hbf[(size_t)(t & 1) * (16 * 512) + b * 512 + col0 + cc] = (_Float16)hn;
            }
        }
        __threadfence();
        __syncthreads();
        if (tid == 0) {
            atomicAdd(cnt, 1u);
            unsigned target = 8u * (unsigned)(t + 1);
            while (__hip_atomic_load(cnt, __ATOMIC_ACQUIRE, __HIP_MEMORY_SCOPE_AGENT) < target)
                __builtin_amdgcn_s_sleep(2);
        }
        __syncthreads();

        if (t + 1 < TT_) {   // refresh full h into LDS for next step (16 KB from L2)
            int row = tid >> 4, colh = (tid & 15) * 32;
            const uint4* src = (const uint4*)(Hbf + (size_t)(t & 1) * (16 * 512) + row * 512 + colh);
            uint4* dst = (uint4*)&hstage[row][colh];
#if HAS_ASYNC_LDS
            async_g2l_b128(src + 0, dst + 0);
            async_g2l_b128(src + 1, dst + 1);
            async_g2l_b128(src + 2, dst + 2);
            async_g2l_b128(src + 3, dst + 3);
            ASYNC_DRAIN();
#else
            dst[0] = src[0]; dst[1] = src[1]; dst[2] = src[2]; dst[3] = src[3];
#endif
            __syncthreads();
        }
    }
}

// ---------------- LayerNorm + gather into f16 ----------------
__global__ __launch_bounds__(256)
void k_ln(const float* __restrict__ HS, const float* __restrict__ ln_g,
          const float* __restrict__ ln_b, _Float16* __restrict__ lnbuf) {
    int bt = blockIdx.x;                   // row m = b*T + t
    int b = bt / T_, t = bt % T_;
    int tid = threadIdx.x;
    __shared__ float2 red[256];

    float vals[4]; float s = 0.f, sq = 0.f;
#pragma unroll
    for (int i = 0; i < 4; ++i) {
        int d = tid + i * 256;
        float v = (d < 512) ? HS[((size_t)t * 16 + b) * 512 + d]
                            : HS[((size_t)(T_ + t) * 16 + (15 - b)) * 512 + (d - 512)];
        vals[i] = v; s += v; sq += v * v;
    }
    red[tid] = make_float2(s, sq);
    __syncthreads();
    for (int off = 128; off > 0; off >>= 1) {
        if (tid < off) { red[tid].x += red[tid + off].x; red[tid].y += red[tid + off].y; }
        __syncthreads();
    }
    float mu  = red[0].x * (1.f / 1024.f);
    float var = red[0].y * (1.f / 1024.f) - mu * mu;
    float rstd = rsqrtf(var + 1e-5f);
#pragma unroll
    for (int i = 0; i < 4; ++i) {
        int d = tid + i * 256;
        float o = (vals[i] - mu) * rstd * ln_g[d] + ln_b[d];
        lnbuf[(size_t)bt * 1024 + d] = (_Float16)o;
    }
}

// ---------------- GEMM 2: projection (16000 x 1024, K=1024) + bias + tanh ----------------
__global__ __launch_bounds__(256, 1)
void k_gemm2(const _Float16* __restrict__ Ag, const _Float16* __restrict__ Wg,
             const float* __restrict__ bias, float* __restrict__ out) {
    __shared__ _Float16 As[128][40];
    __shared__ _Float16 Bs[128][40];
    const int bm = blockIdx.x, bn = blockIdx.y;
    const int tid = threadIdx.x, lane = tid & 31, wave = tid >> 5;
    const int mw = wave >> 1, nw = wave & 1;
    const int kh = lane >> 4, ln15 = lane & 15;

    v8f zero = {0.f,0.f,0.f,0.f,0.f,0.f,0.f,0.f};
    v8f acc[2][4];
#pragma unroll
    for (int i = 0; i < 2; ++i)
#pragma unroll
        for (int j = 0; j < 4; ++j) acc[i][j] = zero;

    const int lr = tid >> 1, lp = (tid & 1) * 16;
    const _Float16* agp = Ag + (size_t)(bm * 128 + lr) * 1024 + lp;
    const _Float16* bgp = Wg + (size_t)(bn * 128 + lr) * 1024 + lp;

    for (int k0 = 0; k0 < 1024; k0 += 32) {
        const uint4* asrc = (const uint4*)(agp + k0);
        const uint4* bsrc = (const uint4*)(bgp + k0);
#if HAS_ASYNC_LDS
        async_g2l_b128(asrc + 0, &As[lr][lp]);
        async_g2l_b128(asrc + 1, &As[lr][lp + 8]);
        async_g2l_b128(bsrc + 0, &Bs[lr][lp]);
        async_g2l_b128(bsrc + 1, &Bs[lr][lp + 8]);
#else
        *(uint4*)&As[lr][lp]     = asrc[0];
        *(uint4*)&As[lr][lp + 8] = asrc[1];
        *(uint4*)&Bs[lr][lp]     = bsrc[0];
        *(uint4*)&Bs[lr][lp + 8] = bsrc[1];
#endif
        if (k0 + 32 < 1024) {
            __builtin_prefetch(agp + k0 + 32, 0, 1);
            __builtin_prefetch(bgp + k0 + 32, 0, 1);
        }
        ASYNC_DRAIN();
        __syncthreads();

        HFrag a[2], b[4];
#pragma unroll
        for (int i = 0; i < 2; ++i) {
            const char* p = (const char*)As + (size_t)(mw * 32 + i * 16 + ln15) * 80;
            a[i].u[0] = *(const uint4*)(p + kh * 16);
            a[i].u[1] = *(const uint4*)(p + 32 + kh * 16);
        }
#pragma unroll
        for (int j = 0; j < 4; ++j) {
            const char* p = (const char*)Bs + (size_t)(nw * 64 + j * 16 + ln15) * 80;
            b[j].u[0] = *(const uint4*)(p + kh * 16);
            b[j].u[1] = *(const uint4*)(p + 32 + kh * 16);
        }
#pragma unroll
        for (int i = 0; i < 2; ++i)
#pragma unroll
            for (int j = 0; j < 4; ++j)
                acc[i][j] = __builtin_amdgcn_wmma_f32_16x16x32_f16(
                    false, a[i].h, false, b[j].h, (short)0, acc[i][j], false, false);
        __syncthreads();
    }
#pragma unroll
    for (int i = 0; i < 2; ++i)
#pragma unroll
        for (int j = 0; j < 4; ++j) {
            int n = bn * 128 + nw * 64 + j * 16 + ln15;
            float bb = bias[n];
#pragma unroll
            for (int r = 0; r < 8; ++r) {
                int m = bm * 128 + mw * 32 + i * 16 + kh * 8 + r;
                out[(size_t)m * 1024 + n] = tanhf(acc[i][j][r] + bb);
            }
        }
}

__global__ void k_xlen(const int* __restrict__ xl, int* __restrict__ dst) {
    if (threadIdx.x < 16) dst[threadIdx.x] = xl[threadIdx.x];
}

// ---------------- launch ----------------
extern "C" void kernel_launch(void* const* d_in, const int* in_sizes, int n_in,
                              void* d_out, int out_size, void* d_ws, size_t ws_size,
                              hipStream_t stream) {
    const float* x     = (const float*)d_in[0];
    const int*   x_len = (const int*)  d_in[1];
    const float* Wz    = (const float*)d_in[2];
    const float* bz    = (const float*)d_in[3];
    const float* Wh    = (const float*)d_in[4];
    const float* bh    = (const float*)d_in[5];
    const float* U     = (const float*)d_in[6];
    const float* zg    = (const float*)d_in[7];
    const float* zb    = (const float*)d_in[8];
    const float* zm    = (const float*)d_in[9];
    const float* zv    = (const float*)d_in[10];
    const float* hg    = (const float*)d_in[11];
    const float* hb    = (const float*)d_in[12];
    const float* hm    = (const float*)d_in[13];
    const float* hv    = (const float*)d_in[14];
    const float* ln_g  = (const float*)d_in[15];
    const float* ln_b  = (const float*)d_in[16];
    const float* pj_W  = (const float*)d_in[17];
    const float* pj_b  = (const float*)d_in[18];

    char* ws = (char*)d_ws;
    _Float16* xc    = (_Float16*)(ws + OFF_XC);
    _Float16* Wc    = (_Float16*)(ws + OFF_WC);
    float*    scale = (float*)   (ws + OFF_SCALE);
    float*    shift = (float*)   (ws + OFF_SHIFT);
    _Float16* U16   = (_Float16*)(ws + OFF_U16);
    _Float16* pjW16 = (_Float16*)(ws + OFF_PJW);
    _Float16* WZp   = (_Float16*)(ws + OFF_WZ);
    _Float16* WHp   = (_Float16*)(ws + OFF_WH);
    float*    HS    = (float*)   (ws + OFF_HS);
    _Float16* lnbuf = (_Float16*)(ws + OFF_LN);
    _Float16* Hbf   = (_Float16*)(ws + OFF_HBF);
    unsigned* cnt   = (unsigned*)(ws + OFF_CNT);

    k_prep_wc   <<<2048, 256, 0, stream>>>(Wz, Wh, Wc);
    k_prep_scale<<<4,    256, 0, stream>>>(bz, bh, zg, zb, zm, zv, hg, hb, hm, hv, scale, shift);
    k_f32_to_f16<<<2048, 256, 0, stream>>>(U, U16, 1024 * 512);
    k_f32_to_f16<<<4096, 256, 0, stream>>>(pj_W, pjW16, 1024 * 1024);
    k_prep_xc   <<<64000,256, 0, stream>>>(x, xc);

    dim3 g1(M1_ / 128, 1024 / 128);
    k_gemm1<<<g1, 256, 0, stream>>>(xc, Wc, scale, shift, WZp, WHp);

    k_zero<<<1, 1, 0, stream>>>(cnt);
    k_rnn <<<8, 256, 0, stream>>>(U16, WZp, WHp, HS, Hbf, cnt);

    k_ln<<<M2_, 256, 0, stream>>>(HS, ln_g, ln_b, lnbuf);

    dim3 g2(M2_ / 128, 1024 / 128);
    k_gemm2<<<g2, 256, 0, stream>>>(lnbuf, pjW16, pj_b, (float*)d_out);

    k_xlen<<<1, 32, 0, stream>>>(x_len, (int*)d_out + (size_t)M2_ * 1024);
}
